// MLCPCLoss_64733747085907
// MI455X (gfx1250) — matrix operand order repaired and checked by reading, not compiled, not run on hardware
//
#include <hip/hip_runtime.h>
#include <hip/hip_bf16.h>

typedef __attribute__((ext_vector_type(16))) _Float16 v16h;
typedef __attribute__((ext_vector_type(8)))  float    v8f;

#define FEAT_DIM 128
#define NCE_K    4096
#define KTOT     (NCE_K + 1)        // 4097
#define NTILES   ((KTOT + 15) / 16) // 257
#define BATCH    256
#define N_FEATS  4
#define N_DATA_F 100000.0f
#define NACC     15                 // S_t,P_t,L_t, S_s[4],P_s[4],L_s[4]
#define NCE_T_INV (1.0f / 0.07f)

// Convert 4 float4 (16 fp32) into one v16h A-chunk (order matches the
// f16 16x16x32 WMMA A layout for this lane's half).
__device__ __forceinline__ v16h cvt_chunk(const float4 x0, const float4 x1,
                                          const float4 x2, const float4 x3) {
    v16h a;
    a[0]  = (_Float16)x0.x; a[1]  = (_Float16)x0.y; a[2]  = (_Float16)x0.z; a[3]  = (_Float16)x0.w;
    a[4]  = (_Float16)x1.x; a[5]  = (_Float16)x1.y; a[6]  = (_Float16)x1.z; a[7]  = (_Float16)x1.w;
    a[8]  = (_Float16)x2.x; a[9]  = (_Float16)x2.y; a[10] = (_Float16)x2.z; a[11] = (_Float16)x2.w;
    a[12] = (_Float16)x3.x; a[13] = (_Float16)x3.y; a[14] = (_Float16)x3.z; a[15] = (_Float16)x3.w;
    return a;
}

__global__ __launch_bounds__(256) void mlcpc_partial(
    const float* __restrict__ feats_s,       // [4,256,128]
    const float* __restrict__ f_t,           // [256,128]
    const int*   __restrict__ idx,           // [256]
    const int*   __restrict__ contrast_idx,  // [256,4097]
    const float* __restrict__ memory_s,      // [100000,128]
    const float* __restrict__ memory_t,      // [100000,128]
    float* __restrict__ partials)            // [256, NACC]
{
    const int b       = blockIdx.x;
    const int lane    = threadIdx.x & 31;
    const int wave    = threadIdx.x >> 5;
    const int halfsel = lane >> 4;   // 0: lanes 0-15, 1: lanes 16-31
    const int lmod    = lane & 15;
    const int myN     = lmod;        // this lane's B/D column

    __shared__ float red[3][5][16];  // [S|P|L][N][wave*2+halfsel]

    // ---- B matrix (loop-invariant, 4 chunks of 32x16 f16) -------------------
    // Column N: 0 -> f_t[b], 1..4 -> feats_s[N-1][b], 5..15 -> zero.
    // f16 B layout (32x16): lanes 0-15 hold col N=lane, K=0..7,16..23 (2 per VGPR);
    // lanes 16-31 hold K=8..15,24..31.
    const float* colptr = nullptr;
    if (myN == 0)       colptr = f_t + (size_t)b * FEAT_DIM;
    else if (myN <= 4)  colptr = feats_s + ((size_t)(myN - 1) * BATCH + b) * FEAT_DIM;

    v16h Bc[4];
    #pragma unroll
    for (int c = 0; c < 4; ++c) {
        v16h bb;
        const int dbase = c * 32 + (halfsel ? 8 : 0);
        if (colptr) {
            #pragma unroll
            for (int j = 0; j < 16; ++j) bb[j] = (_Float16)colptr[dbase + j + (j >= 8 ? 8 : 0)];
        } else {
            #pragma unroll
            for (int j = 0; j < 16; ++j) bb[j] = (_Float16)0.0f;
        }
        Bc[c] = bb;
    }

    float localS = 0.0f, localP = 0.0f, localL = 0.0f;
    const int aoff = halfsel ? 8 : 0;  // A layout lane half offset

    // ---- initial index (handles the k==0 -> idx[b] positive) ----------------
    int sidx = 0;
    {
        const int k0 = wave * 16 + lmod;
        if (k0 < KTOT)
            sidx = (k0 == 0) ? idx[b] : contrast_idx[(size_t)b * KTOT + k0];
    }

    // ---- k-tile loop: wave w handles tiles w, w+8, ... ----------------------
    for (int t = wave; t < NTILES; t += 8) {
        const float* rowS = memory_s + (size_t)sidx * FEAT_DIM + aoff;
        const float* rowT = memory_t + (size_t)sidx * FEAT_DIM + aoff;

        // Prefetch next tile's gather index (k==0 impossible for t+8 >= 8);
        // overlaps the dependent index->gather chain with this tile's compute.
        int sidx_next = 0;
        {
            const int kn = (t + 8) * 16 + lmod;
            if (kn < KTOT) sidx_next = contrast_idx[(size_t)b * KTOT + kn];
        }

        // ---- batch-issue ALL 32 gathers for this tile (distinct registers) --
        float4 rs[16], rt[16];
        #pragma unroll
        for (int c = 0; c < 4; ++c) {
            rs[c * 4 + 0] = *(const float4*)(rowS + c * 32 + 0);
            rs[c * 4 + 1] = *(const float4*)(rowS + c * 32 + 4);
            rs[c * 4 + 2] = *(const float4*)(rowS + c * 32 + 16);
            rs[c * 4 + 3] = *(const float4*)(rowS + c * 32 + 20);
        }
        #pragma unroll
        for (int c = 0; c < 4; ++c) {
            rt[c * 4 + 0] = *(const float4*)(rowT + c * 32 + 0);
            rt[c * 4 + 1] = *(const float4*)(rowT + c * 32 + 4);
            rt[c * 4 + 2] = *(const float4*)(rowT + c * 32 + 16);
            rt[c * 4 + 3] = *(const float4*)(rowT + c * 32 + 20);
        }

        // ---- S chain: only needs the first 16 loads (partial waits) ---------
        v8f Cs = {};
        #pragma unroll
        for (int c = 0; c < 4; ++c) {
            const v16h As = cvt_chunk(rs[c * 4 + 0], rs[c * 4 + 1],
                                      rs[c * 4 + 2], rs[c * 4 + 3]);
            Cs = __builtin_amdgcn_wmma_f32_16x16x32_f16(false, As, false, Bc[c],
                                                        (short)0, Cs, false, false);
        }
        // ---- T chain --------------------------------------------------------
        v8f Ct = {};
        #pragma unroll
        for (int c = 0; c < 4; ++c) {
            const v16h At = cvt_chunk(rt[c * 4 + 0], rt[c * 4 + 1],
                                      rt[c * 4 + 2], rt[c * 4 + 3]);
            Ct = __builtin_amdgcn_wmma_f32_16x16x32_f16(false, At, false, Bc[c],
                                                        (short)0, Ct, false, false);
        }

        // D layout: lane holds (M = v + 8*halfsel, N = myN) for v = 0..7
        if (myN <= 4) {
            #pragma unroll
            for (int v = 0; v < 8; ++v) {
                const int kk = t * 16 + v + (halfsel ? 8 : 0);
                if (kk < KTOT) {
                    const float d  = (myN == 0) ? Cs[v] : Ct[v];
                    const float sc = d * NCE_T_INV;
                    const float e  = __expf(sc);
                    localS += e;
                    if (kk == 0) { localP = e; localL = sc; }  // positive sample
                }
            }
        }
        sidx = sidx_next;
    }

    // ---- deterministic block reduction (no float atomics) -------------------
    if (myN <= 4) {
        const int slot = wave * 2 + halfsel;  // 16 contributors per column
        red[0][myN][slot] = localS;
        red[1][myN][slot] = localP;
        red[2][myN][slot] = localL;
    }
    __syncthreads();
    if (threadIdx.x < 3 * 5) {
        const int q = threadIdx.x / 5;   // 0:S 1:P 2:L
        const int N = threadIdx.x % 5;
        float s = 0.0f;
        #pragma unroll
        for (int i = 0; i < 16; ++i) s += red[q][N][i];
        const int o = (N == 0) ? q : (3 + q * 4 + (N - 1));
        partials[(size_t)b * NACC + o] = s;
    }
}

__global__ __launch_bounds__(32) void mlcpc_finalize(
    const float* __restrict__ partials, float* __restrict__ out)
{
    __shared__ float sums[NACC];
    const int tid = threadIdx.x;
    if (tid < NACC) {
        float s = 0.0f;
        for (int b = 0; b < BATCH; ++b) s += partials[(size_t)b * NACC + tid];
        sums[tid] = s;
    }
    __syncthreads();
    if (tid == 0) {
        const float m      = (float)NCE_K;                    // 4096
        const float cntAll = (float)BATCH * (float)KTOT;      // 256*4097
        const float invB   = 1.0f / (float)BATCH;

        const float S_t = sums[0], P_t = sums[1], L_t = sums[2];
        const float z_t = S_t / cntAll * N_DATA_F;
        const float z_s = sums[3] / cntAll * N_DATA_F;        // mean(outs_s[0]) * N_DATA
        const float logzt = __logf(z_t);
        const float logzs = __logf(z_s);

        // loss_t = N_FEATS * ( log(A + B) + log z_t - mean_b(dot_pos/T) )
        const float At = P_t / (z_t * (float)BATCH * m);
        const float Bt = (S_t - P_t) / (z_t * (float)BATCH);
        const float loss_t = (float)N_FEATS * (__logf(At + Bt) + logzt - L_t * invB);

        float loss_s = 0.0f;
        #pragma unroll
        for (int n = 0; n < N_FEATS; ++n) {
            const float Ss = sums[3 + n], Ps = sums[7 + n], Ls = sums[11 + n];
            const float An = Ps / (z_s * (float)BATCH * m);
            const float Bn = (Ss - Ps) / (z_s * (float)BATCH);
            loss_s += __logf(An + Bn) + logzs - Ls * invB;
        }
        out[0] = loss_s + loss_t;
    }
}

extern "C" void kernel_launch(void* const* d_in, const int* in_sizes, int n_in,
                              void* d_out, int out_size, void* d_ws, size_t ws_size,
                              hipStream_t stream) {
    const float* feats_s      = (const float*)d_in[0];
    const float* f_t          = (const float*)d_in[1];
    const int*   idx          = (const int*)d_in[2];
    const int*   contrast_idx = (const int*)d_in[3];
    const float* memory_s     = (const float*)d_in[4];
    const float* memory_t     = (const float*)d_in[5];

    float* partials = (float*)d_ws;            // 256 * 15 floats = 15 KB
    float* out      = (float*)d_out;

    mlcpc_partial<<<BATCH, 256, 0, stream>>>(feats_s, f_t, idx, contrast_idx,
                                             memory_s, memory_t, partials);
    mlcpc_finalize<<<1, 32, 0, stream>>>(partials, out);
}
